// Head_60275571032173
// MI455X (gfx1250) — matrix operand order repaired and checked by reading, not compiled, not run on hardware
//
#include <hip/hip_runtime.h>

// ---------------------------------------------------------------------------
// Types / helpers
// ---------------------------------------------------------------------------
typedef __attribute__((ext_vector_type(16))) __bf16  v16bf;
typedef __attribute__((ext_vector_type(8)))  float   v8f;
typedef __attribute__((ext_vector_type(4)))  unsigned int u32x4;
typedef __attribute__((ext_vector_type(4)))  float   f32x4;

union Frag16 { v16bf v; u32x4 u[2]; };

static __device__ __forceinline__ unsigned short f2bf(float f) {
    unsigned int u = __float_as_uint(f);
    unsigned int r = u + 0x7FFFu + ((u >> 16) & 1u);   // round to nearest even
    return (unsigned short)(r >> 16);
}

#define SWZ_XOR(v, m) \
    __int_as_float(__builtin_amdgcn_ds_swizzle(__float_as_int(v), (((m) << 10) | 0x1F)))

// LDS byte offset of a __shared__ object (generic LDS address: low 32 bits
// are the wave-relative LDS offset).
static __device__ __forceinline__ unsigned lds_off(const void* p) {
    return (unsigned)(size_t)p;
}

// Async global -> LDS copy, 16B per lane, ASYNCcnt-tracked.
// GVS mode: mem = SADDR(base) + VADDR(voff) + offset; LDS = VDST(ldso) + offset.
#define ASYNC_LD_B128(ldso, base, voff, OFF)                                  \
    asm volatile("global_load_async_to_lds_b128 %0, %1, %2 offset:" #OFF     \
                 :: "v"(ldso), "v"(voff), "s"(base) : "memory")

static __device__ __forceinline__ void wait_async0() {
    asm volatile("s_wait_asynccnt 0x0" ::: "memory");
}

// Problem constants
constexpr int B_  = 8;
constexpr int T_  = 2048;
constexpr int C_  = 1024;
constexpr int HD_ = 128;

// ---------------------------------------------------------------------------
// Kernel 0a: convert x to bf16 (done once; x is compute-reused many times).
// ---------------------------------------------------------------------------
__global__ __launch_bounds__(256) void xbf_kernel(const float* __restrict__ x,
                                                  unsigned short* __restrict__ xbf) {
    size_t base = ((size_t)blockIdx.x * 256 + threadIdx.x) * 8;
    f32x4 a = *(const f32x4*)(x + base);
    f32x4 c = *(const f32x4*)(x + base + 4);
    u32x4 o;
    o.x = (unsigned)f2bf(a.x) | ((unsigned)f2bf(a.y) << 16);
    o.y = (unsigned)f2bf(a.z) | ((unsigned)f2bf(a.w) << 16);
    o.z = (unsigned)f2bf(c.x) | ((unsigned)f2bf(c.y) << 16);
    o.w = (unsigned)f2bf(c.z) | ((unsigned)f2bf(c.w) << 16);
    *(u32x4*)(xbf + base) = o;
}

// ---------------------------------------------------------------------------
// Kernel 0b: transpose + bf16-convert weights. Wt[which][n(128)][k(1024)]
// ---------------------------------------------------------------------------
__global__ __launch_bounds__(256) void wt_kernel(const float* __restrict__ Wq,
                                                 const float* __restrict__ Wk,
                                                 const float* __restrict__ Wv,
                                                 unsigned short* __restrict__ Wt) {
    int idx = blockIdx.x * 256 + threadIdx.x;           // 3*128*1024 threads
    int w   = idx >> 17;
    int rem = idx & 131071;
    int n   = rem >> 10;
    int kk  = rem & 1023;
    const float* W = (w == 0) ? Wq : ((w == 1) ? Wk : Wv);
    Wt[idx] = f2bf(W[kk * HD_ + n]);
}

// ---------------------------------------------------------------------------
// Kernel 1: projection with RoPE, double-buffered async tile staging.
//   gridDim = (128, 3): 128-row block, y = which (0=q,1=k,2=v)
// ---------------------------------------------------------------------------
__global__ __launch_bounds__(256) void proj_kernel(const unsigned short* __restrict__ xbf,
                                                   const unsigned short* __restrict__ Wt,
                                                   const float* __restrict__ cosT,
                                                   const float* __restrict__ sinT,
                                                   unsigned short* __restrict__ qws,
                                                   unsigned short* __restrict__ kws,
                                                   unsigned short* __restrict__ vTws) {
    const int which = blockIdx.y;
    const int m0    = blockIdx.x * 128;
    const int tid   = threadIdx.x;
    const int wid   = tid >> 5;
    const int lane  = tid & 31;
    const int l15   = lane & 15;
    const int lhi   = lane >> 4;

    __shared__ __align__(16) unsigned short As[2][128 * 40];
    __shared__ __align__(16) unsigned short Bs[2][128 * 40];

    v8f acc[8];
#pragma unroll
    for (int j = 0; j < 8; ++j)
#pragma unroll
        for (int i = 0; i < 8; ++i) acc[j][i] = 0.0f;

    const unsigned short* Wtw = Wt + (size_t)which * 131072;

    const int rowA = tid >> 1;
    const int hbA  = (tid & 1) * 16;

    auto issue_tiles = [&](int buf, int k0) {
        {   // A tile: 128 rows x 32 halves of xbf
            unsigned voff = (unsigned)((((m0 + rowA) * C_) + k0 + hbA) * 2);
            unsigned lo   = lds_off(&As[buf][rowA * 40 + hbA]);
            ASYNC_LD_B128(lo, xbf, voff, 0);
            ASYNC_LD_B128(lo, xbf, voff, 16);
        }
        {   // B tile: 128 n-rows x 32 halves of Wt
            unsigned voff = (unsigned)(((rowA * C_) + k0 + hbA) * 2);
            unsigned lo   = lds_off(&Bs[buf][rowA * 40 + hbA]);
            ASYNC_LD_B128(lo, Wtw, voff, 0);
            ASYNC_LD_B128(lo, Wtw, voff, 16);
        }
    };

    issue_tiles(0, 0);

    for (int kb = 0; kb < 32; ++kb) {
        const int buf = kb & 1;
        wait_async0();
        __syncthreads();
        if (kb + 1 < 32) issue_tiles(buf ^ 1, (kb + 1) * 32);

        Frag16 af;
        {
            const unsigned short* p = &As[buf][(16 * wid + l15) * 40 + lhi * 8];
            af.u[0] = *(const u32x4*)(p);
            af.u[1] = *(const u32x4*)(p + 16);
        }
#pragma unroll
        for (int j = 0; j < 8; ++j) {
            Frag16 bf_;
            const unsigned short* p = &Bs[buf][(16 * j + l15) * 40 + lhi * 8];
            bf_.u[0] = *(const u32x4*)(p);
            bf_.u[1] = *(const u32x4*)(p + 16);
            acc[j] = __builtin_amdgcn_wmma_f32_16x16x32_bf16(
                false, af.v, false, bf_.v, (short)0, acc[j], false, false);
        }
    }

    // Epilogue: RoPE (q,k), 1/sqrt(128) folded into q, convert + store.
    const float qscale = 0.08838834764831845f;
#pragma unroll
    for (int j = 0; j < 8; ++j) {
        int n = 16 * j + l15;
#pragma unroll
        for (int r = 0; r < 8; ++r) {
            int   m = m0 + 16 * wid + 8 * lhi + r;
            int   t = m & (T_ - 1);
            float val = acc[j][r];
            if (which < 2) {
                float partner = SWZ_XOR(val, 1);        // lane^1 holds paired column
                int   d = n >> 1;
                float c = cosT[t * 64 + d];
                float s = sinT[t * 64 + d];
                val = (lane & 1) ? (val * c + partner * s)    // odd:  xe*s + xo*c
                                 : (val * c - partner * s);   // even: xe*c - xo*s
            }
            if (which == 0) {
                qws[(size_t)m * HD_ + n] = f2bf(val * qscale);
            } else if (which == 1) {
                kws[(size_t)m * HD_ + n] = f2bf(val);
            } else {
                int bb = m >> 11;
                vTws[(((size_t)(bb * HD_ + n)) << 11) + t] = f2bf(val);
            }
        }
    }
}

// ---------------------------------------------------------------------------
// Kernel 2: causal flash attention, double-buffered async K/V^T staging.
//   gridDim = (T/128, B). 8 waves; each wave: 16 query rows, online softmax.
// ---------------------------------------------------------------------------
__global__ __launch_bounds__(256) void attn_kernel(const unsigned short* __restrict__ q,
                                                   const unsigned short* __restrict__ k,
                                                   const unsigned short* __restrict__ vT,
                                                   float* __restrict__ out) {
    const int b     = blockIdx.y;
    const int qbase = blockIdx.x * 128;
    const int tid   = threadIdx.x;
    const int wid   = tid >> 5;
    const int lane  = tid & 31;
    const int l15   = lane & 15;
    const int lhi   = lane >> 4;

    __shared__ __align__(16) unsigned short Ks [2][32 * 136];  // [key][d]
    __shared__ __align__(16) unsigned short VTs[2][128 * 40];  // [d][key]
    __shared__ __align__(16) unsigned short Ps [8 * 16 * 40];  // per-wave P patch

    // q fragments (4 K-chunks of 32), loaded once.
    Frag16 qa[4];
    {
        const unsigned short* qrow =
            q + (size_t)(b * T_ + qbase + 16 * wid + l15) * HD_;
#pragma unroll
        for (int c = 0; c < 4; ++c) {
            const unsigned short* p = qrow + c * 32 + lhi * 8;
            qa[c].u[0] = *(const u32x4*)(p);
            qa[c].u[1] = *(const u32x4*)(p + 16);
        }
    }

    v8f   O[8];
    float mrow[8], lrow[8];
#pragma unroll
    for (int j = 0; j < 8; ++j)
#pragma unroll
        for (int i = 0; i < 8; ++i) O[j][i] = 0.0f;
#pragma unroll
    for (int r = 0; r < 8; ++r) { mrow[r] = -__builtin_inff(); lrow[r] = 0.0f; }

    auto issue_kv = [&](int buf, int key0) {
        {   // K tile: 32 keys x 128 halves
            int row = tid >> 3, ch = tid & 7;
            unsigned voff = (unsigned)((((b * T_) + key0 + row) * HD_ + ch * 16) * 2);
            unsigned lo   = lds_off(&Ks[buf][row * 136 + ch * 16]);
            ASYNC_LD_B128(lo, k, voff, 0);
            ASYNC_LD_B128(lo, k, voff, 16);
        }
        {   // V^T tile: 128 d-rows x 32 halves
            int d = tid >> 1, hh = (tid & 1) * 16;
            unsigned voff = (unsigned)((((((b * HD_) + d)) << 11) + key0 + hh) * 2);
            unsigned lo   = lds_off(&VTs[buf][d * 40 + hh]);
            ASYNC_LD_B128(lo, vT, voff, 0);
            ASYNC_LD_B128(lo, vT, voff, 16);
        }
    };

    const int nIter = (qbase + 128) >> 5;                // causal key-tile count
    issue_kv(0, 0);

    for (int it = 0; it < nIter; ++it) {
        const int buf  = it & 1;
        const int key0 = it << 5;
        wait_async0();
        __syncthreads();
        if (it + 1 < nIter) issue_kv(buf ^ 1, key0 + 32);

        // S = q . k^T  (2 key-subtiles x 4 K-chunks)
        v8f S[2];
#pragma unroll
        for (int j = 0; j < 2; ++j)
#pragma unroll
            for (int i = 0; i < 8; ++i) S[j][i] = 0.0f;
#pragma unroll
        for (int c = 0; c < 4; ++c) {
#pragma unroll
            for (int j = 0; j < 2; ++j) {
                Frag16 kb;
                const unsigned short* p = &Ks[buf][(16 * j + l15) * 136 + c * 32 + lhi * 8];
                kb.u[0] = *(const u32x4*)(p);
                kb.u[1] = *(const u32x4*)(p + 16);
                S[j] = __builtin_amdgcn_wmma_f32_16x16x32_bf16(
                    false, qa[c].v, false, kb.v, (short)0, S[j], false, false);
            }
        }

        // Causal mask + online softmax update.
#pragma unroll
        for (int r = 0; r < 8; ++r) {
            int m = qbase + 16 * wid + 8 * lhi + r;
#pragma unroll
            for (int j = 0; j < 2; ++j) {
                int key = key0 + 16 * j + l15;
                if (key > m) S[j][r] = -__builtin_inff();
            }
            float v = fmaxf(S[0][r], S[1][r]);
            v = fmaxf(v, SWZ_XOR(v, 1));
            v = fmaxf(v, SWZ_XOR(v, 2));
            v = fmaxf(v, SWZ_XOR(v, 4));
            v = fmaxf(v, SWZ_XOR(v, 8));
            float mnew  = fmaxf(mrow[r], v);
            float alpha = __expf(mrow[r] - mnew);
            mrow[r] = mnew;
            lrow[r] *= alpha;
#pragma unroll
            for (int j2 = 0; j2 < 8; ++j2) O[j2][r] *= alpha;
            float rs = 0.0f;
#pragma unroll
            for (int j = 0; j < 2; ++j) {
                float p = __expf(S[j][r] - mnew);
                S[j][r] = p;
                rs += p;
            }
            rs += SWZ_XOR(rs, 1);
            rs += SWZ_XOR(rs, 2);
            rs += SWZ_XOR(rs, 4);
            rs += SWZ_XOR(rs, 8);
            lrow[r] += rs;
        }

        // Re-layout P (C-frag -> A-frag) through wave-private LDS patch.
        // Same-wave LDS ops are architecturally in-order: no barrier needed.
        {
            unsigned short* Pw = Ps + wid * (16 * 40);
#pragma unroll
            for (int r = 0; r < 8; ++r)
#pragma unroll
                for (int j = 0; j < 2; ++j)
                    Pw[(8 * lhi + r) * 40 + 16 * j + l15] = f2bf(S[j][r]);
        }

        // O += P @ V
        Frag16 pa;
        {
            const unsigned short* Pr = Ps + wid * (16 * 40) + l15 * 40 + lhi * 8;
            pa.u[0] = *(const u32x4*)(Pr);
            pa.u[1] = *(const u32x4*)(Pr + 16);
        }
#pragma unroll
        for (int j2 = 0; j2 < 8; ++j2) {
            Frag16 vb;
            const unsigned short* p = &VTs[buf][(16 * j2 + l15) * 40 + lhi * 8];
            vb.u[0] = *(const u32x4*)(p);
            vb.u[1] = *(const u32x4*)(p + 16);
            O[j2] = __builtin_amdgcn_wmma_f32_16x16x32_bf16(
                false, pa.v, false, vb.v, (short)0, O[j2], false, false);
        }
    }

    // Epilogue: divide by row sum, write fp32 output.
#pragma unroll
    for (int r = 0; r < 8; ++r) {
        float inv = 1.0f / lrow[r];
        int   m   = qbase + 16 * wid + 8 * lhi + r;
        float* orow = out + (size_t)(b * T_ + m) * HD_;
#pragma unroll
        for (int j2 = 0; j2 < 8; ++j2)
            orow[16 * j2 + l15] = O[j2][r] * inv;
    }
}

// ---------------------------------------------------------------------------
// Launch
// ---------------------------------------------------------------------------
extern "C" void kernel_launch(void* const* d_in, const int* in_sizes, int n_in,
                              void* d_out, int out_size, void* d_ws, size_t ws_size,
                              hipStream_t stream) {
    const float* x    = (const float*)d_in[0];
    const float* cosT = (const float*)d_in[1];
    const float* sinT = (const float*)d_in[2];
    const float* Wk   = (const float*)d_in[3];
    const float* Wq   = (const float*)d_in[4];
    const float* Wv   = (const float*)d_in[5];
    float* out = (float*)d_out;

    char* ws = (char*)d_ws;
    unsigned short* Wt   = (unsigned short*)(ws + 0);           //  0.75 MB
    unsigned short* xbf  = (unsigned short*)(ws + 786432);      // 32 MB
    unsigned short* qws  = (unsigned short*)(ws + 34340864);    //  4 MB
    unsigned short* kws  = (unsigned short*)(ws + 38535168);    //  4 MB
    unsigned short* vTws = (unsigned short*)(ws + 42729472);    //  4 MB

    xbf_kernel <<<8192, 256, 0, stream>>>(x, xbf);
    wt_kernel  <<<1536, 256, 0, stream>>>(Wq, Wk, Wv, Wt);
    proj_kernel<<<dim3(128, 3), 256, 0, stream>>>(xbf, Wt, cosT, sinT, qws, kws, vTws);
    attn_kernel<<<dim3(T_ / 128, B_), 256, 0, stream>>>(qws, kws, vTws, out);
}
